// HybridMambaBlock_49074296324241
// MI455X (gfx1250) — matrix operand order repaired and checked by reading, not compiled, not run on hardware
//
#include <hip/hip_runtime.h>
#include <hip/hip_bf16.h>

// ---------------------------------------------------------------------------
// HybridMambaBlock for MI455X (gfx1250, wave32, WMMA).
//   1. cvt: W_in / W_res / W_out  f32 -> f16
//   2. ln:  LayerNorm(x) -> xn_f16 ; also x -> x_f16
//   3. WMMA GEMM: p   = xn_f16 @ W_in_f16^T   (2048 x 52272, K=768)  <- 94% FLOPs
//   4. WMMA GEMM: res = x_f16  @ W_res_f16^T  (2048 x 1536,  K=768)
//   5. dt  = clip(softplus(dth @ W_dt^T + bias))
//   6. u   = causal depthwise conv(K=4) + bias
//   7. selective scan: 1 thread per (b,d), 16 f32 states in registers
//   8. gate: g = y*silu(z); RMSNorm; + res -> g_f16
//   9. WMMA GEMM: out = g_f16 @ W_out_f16^T  (2048 x 768, K=1536) -> d_out
// GEMM: block = 8 waves; A tile (16 x K) staged to LDS via async-to-LDS path;
// each wave owns a 16x64 strip (4 accumulators) -> 4 WMMAs per A fragment.
// ---------------------------------------------------------------------------

typedef __attribute__((ext_vector_type(16))) _Float16 v16h;
typedef __attribute__((ext_vector_type(8)))  _Float16 v8h;
typedef __attribute__((ext_vector_type(8)))  float    v8f;
typedef __attribute__((ext_vector_type(4)))  float    v4f;
typedef __attribute__((ext_vector_type(4)))  int      v4i;

#define D_MODEL 768
#define D_INNER 1536
#define D_STATE 16
#define D_CONV  4
#define DT_RANK 48
#define BATCH   2
#define SEQLEN  1024
#define ROWS    (BATCH * SEQLEN)                       // 2048
#define P_N     (2 * D_INNER + DT_RANK + 2 * D_INNER * D_STATE) // 52272
#define OFF_Z   0
#define OFF_U   D_INNER                                // 1536
#define OFF_DTH (2 * D_INNER)                          // 3072
#define OFF_B   (2 * D_INNER + DT_RANK)                // 3120
#define OFF_C   (OFF_B + D_INNER * D_STATE)            // 27696
#define DT_MIN_F 1e-4f
#define DT_MAX_F 1.0f

#define GEMM_WAVES   8     // waves per block
#define NT_PER_WAVE  4     // 16-wide N tiles per wave (16x64 strip)
#define NT_PER_BLOCK (GEMM_WAVES * NT_PER_WAVE)  // 32 tiles = 512 columns

#if defined(__AMDGCN__)
#if __has_builtin(__builtin_amdgcn_global_load_async_to_lds_b128)
#define HAS_ASYNC_LDS 1
// Builtin takes 128-bit int-vector pointers: AS1 (global) src, AS3 (LDS) dst.
typedef __attribute__((address_space(1))) v4i gas_v4i;
typedef __attribute__((address_space(3))) v4i las_v4i;
#endif
#endif

__device__ inline void wait_asynccnt0() {
#if defined(__AMDGCN__)
#if __has_builtin(__builtin_amdgcn_s_wait_asynccnt)
  __builtin_amdgcn_s_wait_asynccnt(0);
#else
  asm volatile("s_wait_asynccnt 0x0" ::: "memory");
#endif
#endif
}

// ----------------------------- f32 -> f16 conversion ------------------------
__global__ void cvt_f32_to_f16_kernel(const float* __restrict__ src,
                                      _Float16* __restrict__ dst, long long n) {
  long long i = (long long)blockIdx.x * blockDim.x + threadIdx.x;
  long long stride = (long long)gridDim.x * blockDim.x;
  for (; i < n; i += stride) dst[i] = (_Float16)src[i];
}

// ----------------------------- LayerNorm ------------------------------------
__global__ void ln_kernel(const float* __restrict__ x,
                          const float* __restrict__ lnw,
                          const float* __restrict__ lnb,
                          _Float16* __restrict__ xn_h,
                          _Float16* __restrict__ x_h) {
  __shared__ float red[256];
  const int row = blockIdx.x;
  const int tid = threadIdx.x;
  const float* xr = x + (size_t)row * D_MODEL;

  float s = 0.f;
  for (int i = tid; i < D_MODEL; i += 256) s += xr[i];
  red[tid] = s; __syncthreads();
  for (int st = 128; st > 0; st >>= 1) {
    if (tid < st) red[tid] += red[tid + st];
    __syncthreads();
  }
  const float mu = red[0] * (1.f / D_MODEL);
  __syncthreads();

  float v = 0.f;
  for (int i = tid; i < D_MODEL; i += 256) { float d = xr[i] - mu; v += d * d; }
  red[tid] = v; __syncthreads();
  for (int st = 128; st > 0; st >>= 1) {
    if (tid < st) red[tid] += red[tid + st];
    __syncthreads();
  }
  const float rstd = rsqrtf(red[0] * (1.f / D_MODEL) + 1e-5f);

  for (int i = tid; i < D_MODEL; i += 256) {
    float xv = xr[i];
    xn_h[(size_t)row * D_MODEL + i] = (_Float16)((xv - mu) * rstd * lnw[i] + lnb[i]);
    x_h [(size_t)row * D_MODEL + i] = (_Float16)xv;
  }
}

// ----------------------------- WMMA GEMM (NT) --------------------------------
// C[M,N] = A[M,K] * W[N,K]^T.  A tile (16 x K halves) is staged into LDS once
// per block (async-to-LDS when available); each wave computes a 16x64 strip.
__global__ __launch_bounds__(GEMM_WAVES * 32)
void wmma_gemm_nt_kernel(const _Float16* __restrict__ A,
                         const _Float16* __restrict__ W,
                         float* __restrict__ C,
                         int M, int N, int K) {
  extern __shared__ _Float16 sA[];     // 16 * K halves
  const int tid  = threadIdx.x;
  const int lane = tid & 31;
  const int wave = tid >> 5;
  const int m0   = blockIdx.y * 16;

  // ---- Stage A tile (16 rows x K) into LDS. K % 8 == 0 so 16B chunks never
  // cross a row boundary.
  {
    const int elems  = 16 * K;         // halves
    const int chunks = elems / 8;      // 16-byte chunks
    for (int c = tid; c < chunks; c += GEMM_WAVES * 32) {
      const int e = c * 8;
      const int r = e / K, k = e % K;
      const _Float16* gsrc = A + (size_t)(m0 + r) * K + k;
#if defined(HAS_ASYNC_LDS)
      __builtin_amdgcn_global_load_async_to_lds_b128(
          (gas_v4i*)(void*)gsrc,
          (las_v4i*)(void*)(sA + e),
          /*offset=*/0, /*cpol=*/0);
#else
      *(v8h*)(sA + e) = *(const v8h*)gsrc;
#endif
    }
#if defined(HAS_ASYNC_LDS)
    wait_asynccnt0();
#endif
    __syncthreads();
  }

  const int rw  = lane & 15;   // A row within tile / B column within tile
  const int grp = lane >> 4;   // lane-group selects K sub-range

  // Wave-uniform N strip; out-of-range tiles load row 0 (safe) and skip store.
  const int nbase = (blockIdx.x * GEMM_WAVES + wave) * (16 * NT_PER_WAVE);
  const _Float16* wrow[NT_PER_WAVE];
  bool act[NT_PER_WAVE];
#pragma unroll
  for (int t = 0; t < NT_PER_WAVE; ++t) {
    const int n0 = nbase + t * 16;
    act[t]  = (n0 < N);
    wrow[t] = W + (size_t)((act[t] ? n0 : 0) + rw) * K;
  }

  v8f acc[NT_PER_WAVE] = {};
  for (int k0 = 0; k0 < K; k0 += 32) {
    // A 16x32 fragment from LDS: lane-group 0 holds K {0..7,16..23},
    // lane-group 1 holds K {8..15,24..31} (ISA 7.12.2).
    v8h alo = *(const v8h*)(sA + rw * K + k0 + grp * 8);
    v8h ahi = *(const v8h*)(sA + rw * K + k0 + 16 + grp * 8);
    v16h a;
#pragma unroll
    for (int h = 0; h < 8; ++h) { a[h] = alo[h]; a[h + 8] = ahi[h]; }

#pragma unroll
    for (int t = 0; t < NT_PER_WAVE; ++t) {
      __builtin_prefetch(wrow[t] + k0 + 32, 0, 1);
      // B 32x16 fragment: lane-group 0 holds K {0..7,16..23}, group 1 K {8..15,24..31}.
      v8h blo = *(const v8h*)(wrow[t] + k0 + grp * 8);
      v8h bhi = *(const v8h*)(wrow[t] + k0 + 16 + grp * 8);
      v16h b;
#pragma unroll
      for (int h = 0; h < 8; ++h) { b[h] = blo[h]; b[h + 8] = bhi[h]; }
      acc[t] = __builtin_amdgcn_wmma_f32_16x16x32_f16(
          false, a, false, b, (short)0, acc[t], false, false);
    }
  }

  // C/D layout: VGPR i, lanes 0-15 -> M=i ; lanes 16-31 -> M=i+8.
#pragma unroll
  for (int t = 0; t < NT_PER_WAVE; ++t) {
    if (!act[t]) continue;
    const int n0 = nbase + t * 16;
#pragma unroll
    for (int i = 0; i < 8; ++i) {
      const int r = i + 8 * grp;
      C[(size_t)(m0 + r) * N + n0 + rw] = acc[t][i];
    }
  }
}

// ----------------------------- dt projection + softplus ----------------------
__global__ void dt_kernel(const float* __restrict__ p,
                          const float* __restrict__ W_dt,
                          const float* __restrict__ dt_bias,
                          float* __restrict__ dt) {
  int i = blockIdx.x * blockDim.x + threadIdx.x;
  if (i >= ROWS * D_INNER) return;
  const int row = i / D_INNER;
  const int d   = i % D_INNER;
  const float* dth = p + (size_t)row * P_N + OFF_DTH;
  const float* wr  = W_dt + (size_t)d * DT_RANK;
  float s = dt_bias[d];
#pragma unroll 8
  for (int r = 0; r < DT_RANK; ++r) s += dth[r] * wr[r];
  float sp = (s > 20.f) ? s : log1pf(__expf(s));
  dt[i] = fminf(fmaxf(sp, DT_MIN_F), DT_MAX_F);
}

// ----------------------------- causal depthwise conv (K=4) -------------------
__global__ void conv_kernel(const float* __restrict__ p,
                            const float* __restrict__ conv_w,
                            const float* __restrict__ conv_b,
                            float* __restrict__ u) {
  int i = blockIdx.x * blockDim.x + threadIdx.x;
  if (i >= ROWS * D_INNER) return;
  const int d  = i % D_INNER;
  const int bl = i / D_INNER;
  const int l  = bl % SEQLEN;
  const int b  = bl / SEQLEN;
  float s = conv_b[d];
#pragma unroll
  for (int k = 0; k < D_CONV; ++k) {
    int lk = l - (D_CONV - 1) + k;
    if (lk >= 0)
      s += conv_w[d * D_CONV + k] *
           p[(size_t)(b * SEQLEN + lk) * P_N + OFF_U + d];
  }
  u[i] = s;
}

// ----------------------------- selective scan --------------------------------
__global__ void scan_kernel(const float* __restrict__ p,
                            const float* __restrict__ dt,
                            const float* __restrict__ u,
                            const float* __restrict__ A_log,
                            const float* __restrict__ Dskip,
                            float* __restrict__ y) {
  int idx = blockIdx.x * blockDim.x + threadIdx.x;
  if (idx >= BATCH * D_INNER) return;
  const int b = idx / D_INNER;
  const int d = idx % D_INNER;

  float Aval[D_STATE];
#pragma unroll
  for (int n = 0; n < D_STATE; ++n) Aval[n] = -__expf(A_log[d * D_STATE + n]);
  float h[D_STATE];
#pragma unroll
  for (int n = 0; n < D_STATE; ++n) h[n] = 0.f;
  const float dsk = Dskip[d];

  for (int l = 0; l < SEQLEN; ++l) {
    const size_t row = (size_t)(b * SEQLEN + l);
    const float dtv = dt[row * D_INNER + d];
    const float uv  = u[row * D_INNER + d];
    const float dtu = dtv * uv;
    const float* Br = p + row * P_N + OFF_B + d * D_STATE;
    const float* Cr = p + row * P_N + OFF_C + d * D_STATE;
    float acc = 0.f;
#pragma unroll
    for (int q = 0; q < D_STATE / 4; ++q) {
      v4f Bv = *(const v4f*)(Br + q * 4);
      v4f Cv = *(const v4f*)(Cr + q * 4);
#pragma unroll
      for (int j = 0; j < 4; ++j) {
        int n = q * 4 + j;
        float dA = __expf(dtv * Aval[n]);
        h[n] = dA * h[n] + dtu * Bv[j];
        acc += Cv[j] * h[n];
      }
    }
    y[row * D_INNER + d] = acc + uv * dsk;
  }
}

// ----------------------------- gate + RMSNorm + residual ---------------------
__global__ void gate_kernel(const float* __restrict__ y,
                            const float* __restrict__ p,
                            const float* __restrict__ res,
                            const float* __restrict__ norm_w,
                            _Float16* __restrict__ g_h) {
  __shared__ float red[256];
  const int row = blockIdx.x;
  const int tid = threadIdx.x;
  const float* yr = y + (size_t)row * D_INNER;
  const float* zr = p + (size_t)row * P_N + OFF_Z;

  float gloc[D_INNER / 256];
  float s = 0.f;
#pragma unroll
  for (int j = 0; j < D_INNER / 256; ++j) {
    int d = tid + j * 256;
    float z = zr[d];
    float g = yr[d] * (z / (1.f + __expf(-z)));  // y * silu(z)
    gloc[j] = g;
    s += g * g;
  }
  red[tid] = s; __syncthreads();
  for (int st = 128; st > 0; st >>= 1) {
    if (tid < st) red[tid] += red[tid + st];
    __syncthreads();
  }
  const float rr = rsqrtf(red[0] * (1.f / D_INNER) + 1e-6f);

#pragma unroll
  for (int j = 0; j < D_INNER / 256; ++j) {
    int d = tid + j * 256;
    g_h[(size_t)row * D_INNER + d] =
        (_Float16)(gloc[j] * rr * norm_w[d] + res[(size_t)row * D_INNER + d]);
  }
}

// ----------------------------- launch ---------------------------------------
static inline size_t align256(size_t x) { return (x + 255) & ~(size_t)255; }

extern "C" void kernel_launch(void* const* d_in, const int* in_sizes, int n_in,
                              void* d_out, int out_size, void* d_ws, size_t ws_size,
                              hipStream_t stream) {
  (void)in_sizes; (void)n_in; (void)out_size; (void)ws_size;
  const float* x       = (const float*)d_in[0];
  const float* ln_w    = (const float*)d_in[1];
  const float* ln_b    = (const float*)d_in[2];
  const float* W_in    = (const float*)d_in[3];
  const float* W_dt    = (const float*)d_in[4];
  const float* conv_w  = (const float*)d_in[5];
  const float* conv_b  = (const float*)d_in[6];
  const float* A_log   = (const float*)d_in[7];
  const float* Dskip   = (const float*)d_in[8];
  const float* dt_bias = (const float*)d_in[9];
  const float* norm_w  = (const float*)d_in[10];
  const float* W_res   = (const float*)d_in[11];
  const float* W_out   = (const float*)d_in[12];
  float* out = (float*)d_out;

  // Workspace carve-up (256B aligned).
  char* ws = (char*)d_ws;
  size_t off = 0;
  _Float16* W_in_h  = (_Float16*)(ws + off); off = align256(off + (size_t)P_N * D_MODEL * 2);
  _Float16* W_res_h = (_Float16*)(ws + off); off = align256(off + (size_t)D_INNER * D_MODEL * 2);
  _Float16* W_out_h = (_Float16*)(ws + off); off = align256(off + (size_t)D_MODEL * D_INNER * 2);
  _Float16* xn_h    = (_Float16*)(ws + off); off = align256(off + (size_t)ROWS * D_MODEL * 2);
  _Float16* x_h     = (_Float16*)(ws + off); off = align256(off + (size_t)ROWS * D_MODEL * 2);
  float*    pbuf    = (float*)   (ws + off); off = align256(off + (size_t)ROWS * P_N * 4);
  float*    ubuf    = (float*)   (ws + off); off = align256(off + (size_t)ROWS * D_INNER * 4);
  float*    dtbuf   = (float*)   (ws + off); off = align256(off + (size_t)ROWS * D_INNER * 4);
  float*    ybuf    = (float*)   (ws + off); off = align256(off + (size_t)ROWS * D_INNER * 4);
  float*    resbuf  = (float*)   (ws + off); off = align256(off + (size_t)ROWS * D_INNER * 4);
  _Float16* g_h     = (_Float16*)(ws + off); off = align256(off + (size_t)ROWS * D_INNER * 2);

  // 1. weight conversions
  cvt_f32_to_f16_kernel<<<2048, 256, 0, stream>>>(W_in,  W_in_h,  (long long)P_N * D_MODEL);
  cvt_f32_to_f16_kernel<<<512,  256, 0, stream>>>(W_res, W_res_h, (long long)D_INNER * D_MODEL);
  cvt_f32_to_f16_kernel<<<512,  256, 0, stream>>>(W_out, W_out_h, (long long)D_MODEL * D_INNER);

  // 2. LayerNorm
  ln_kernel<<<ROWS, 256, 0, stream>>>(x, ln_w, ln_b, xn_h, x_h);

  const int threads = GEMM_WAVES * 32;
  // 3. p = xn @ W_in^T   (M=2048, N=52272, K=768)
  {
    const int ntiles = (P_N + 15) / 16;  // 3267
    dim3 grid((ntiles + NT_PER_BLOCK - 1) / NT_PER_BLOCK, ROWS / 16);
    wmma_gemm_nt_kernel<<<grid, threads, 16 * D_MODEL * 2, stream>>>(
        xn_h, W_in_h, pbuf, ROWS, P_N, D_MODEL);
  }
  // 4. res = x @ W_res^T (M=2048, N=1536, K=768)
  {
    dim3 grid((D_INNER / 16 + NT_PER_BLOCK - 1) / NT_PER_BLOCK, ROWS / 16);
    wmma_gemm_nt_kernel<<<grid, threads, 16 * D_MODEL * 2, stream>>>(
        x_h, W_res_h, resbuf, ROWS, D_INNER, D_MODEL);
  }

  // 5. dt
  dt_kernel<<<(ROWS * D_INNER + 255) / 256, 256, 0, stream>>>(pbuf, W_dt, dt_bias, dtbuf);
  // 6. conv
  conv_kernel<<<(ROWS * D_INNER + 255) / 256, 256, 0, stream>>>(pbuf, conv_w, conv_b, ubuf);
  // 7. scan
  scan_kernel<<<(BATCH * D_INNER + 255) / 256, 256, 0, stream>>>(pbuf, dtbuf, ubuf, A_log, Dskip, ybuf);
  // 8. gate + RMSNorm + residual
  gate_kernel<<<ROWS, 256, 0, stream>>>(ybuf, pbuf, resbuf, norm_w, g_h);

  // 9. out = g @ W_out^T (M=2048, N=768, K=1536)
  {
    dim3 grid((D_MODEL / 16 + NT_PER_BLOCK - 1) / NT_PER_BLOCK, ROWS / 16);
    wmma_gemm_nt_kernel<<<grid, threads, 16 * D_INNER * 2, stream>>>(
        g_h, W_out_h, out, ROWS, D_MODEL, D_INNER);
  }
}
